// WindowedAttention_5145370821104
// MI455X (gfx1250) — compile-verified, matrix operand order checked by reading, not run on hardware
//
#include <hip/hip_runtime.h>
#include <hip/hip_bf16.h>

// ---------------------------------------------------------------------------
// Windowed attention (Swin-style) for MI455X / gfx1250, wave32 + WMMA bf16.
//   B=8, H=W=60, C=768, NH=12, hd=64, WS=14 -> Hp=Wp=70, 25 windows, Nw=196.
// Pipeline:
//   0) f32 -> bf16 convert of x; convert+transpose of w_qkv / w_proj.
//   1) GEMM1: qkv[28800,2304] = x[28800,768] @ w_qkv      (bf16 WMMA, f32 acc)
//   2) attention: per (b, window, head) block, WMMA qk^T + softmax + WMMA pv
//   3) GEMM2: out[28800,768] = ao[28800,768] @ w_proj + b (f32 output)
// Workspace (~226 MB of d_ws): xb | wqkvT | wprojT | qkvb | aob  (all bf16).
// ---------------------------------------------------------------------------

typedef __attribute__((ext_vector_type(16))) __bf16        v16bf;
typedef __attribute__((ext_vector_type(8)))  float         v8f;
typedef __attribute__((ext_vector_type(4)))  unsigned int  u32x4;

union Frag16 { u32x4 q[2]; v16bf v; };   // one 16x32 (or 32x16) bf16 operand

__device__ __forceinline__ unsigned short f2bf(float f) {
  unsigned int u = __float_as_uint(f);
  u += 0x7FFFu + ((u >> 16) & 1u);          // round-to-nearest-even
  return (unsigned short)(u >> 16);
}

// ---------------------------------------------------------------- converts --
__global__ __launch_bounds__(256) void k_f32_to_bf16(
    const float* __restrict__ in, unsigned short* __restrict__ out, int n) {
  int stride = gridDim.x * blockDim.x;
  for (int i = blockIdx.x * blockDim.x + threadIdx.x; i < n; i += stride)
    out[i] = f2bf(in[i]);
}

// in [K][N] f32  ->  out [N][K] bf16   (B^T layout for K-contiguous frags)
__global__ __launch_bounds__(256) void k_transpose_to_bf16(
    const float* __restrict__ in, unsigned short* __restrict__ out, int K, int N) {
  int stride = gridDim.x * blockDim.x;
  int total = K * N;
  for (int i = blockIdx.x * blockDim.x + threadIdx.x; i < total; i += stride) {
    int kk = i / N, nn = i - kk * N;
    out[(size_t)nn * K + kk] = f2bf(in[i]);
  }
}

// -------------------------------------------------------------- bf16 GEMM --
// C[M,N] = A[M,K] @ B[K,N], A row-major bf16, Bt = B^T row-major bf16.
// Block tile 128x128, 8 waves (2x4), wave tile 64x32, K-step 32.
template <bool OUT_F32, bool ADD_BIAS>
__global__ __launch_bounds__(256) void k_gemm_bf16(
    const unsigned short* __restrict__ A,
    const unsigned short* __restrict__ Bt,
    const float* __restrict__ bias,
    void* __restrict__ Cout, int M, int N, int K) {
  __shared__ unsigned short As[128 * 32];
  __shared__ unsigned short Bs[128 * 32];

  const int tid  = threadIdx.x;
  const int lane = tid & 31, wave = tid >> 5;
  const int wm = wave >> 2, wn = wave & 3;          // 2 x 4 wave grid
  const int bm = blockIdx.y * 128, bn = blockIdx.x * 128;

  v8f acc[4][2];
#pragma unroll
  for (int i = 0; i < 4; ++i)
#pragma unroll
    for (int j = 0; j < 2; ++j)
#pragma unroll
      for (int e = 0; e < 8; ++e) acc[i][j][e] = 0.0f;

  const int lr = tid >> 1;                          // 0..127 tile row
  const int lc = (tid & 1) * 16;                    // 0 / 16 K-chunk

  for (int k0 = 0; k0 < K; k0 += 32) {
    __syncthreads();
    const unsigned short* ga = A  + (size_t)(bm + lr) * K + k0 + lc;
    const unsigned short* gb = Bt + (size_t)(bn + lr) * K + k0 + lc;
    *(u32x4*)&As[lr * 32 + lc]     = *(const u32x4*)ga;
    *(u32x4*)&As[lr * 32 + lc + 8] = *(const u32x4*)(ga + 8);
    *(u32x4*)&Bs[lr * 32 + lc]     = *(const u32x4*)gb;
    *(u32x4*)&Bs[lr * 32 + lc + 8] = *(const u32x4*)(gb + 8);
    if (k0 + 32 < K) {                              // global_prefetch_b8 path
      __builtin_prefetch(ga + 32, 0, 3);
      __builtin_prefetch(gb + 32, 0, 3);
    }
    __syncthreads();

    const int hl = lane >> 4;                       // K half select
    const int mn = lane & 15;                       // M (A) / N (B) index
    Frag16 bf[2];
#pragma unroll
    for (int j = 0; j < 2; ++j) {
      const unsigned short* p = &Bs[(wn * 32 + j * 16 + mn) * 32 + hl * 8];
      bf[j].q[0] = *(const u32x4*)p;                // K = hl*8 .. +7
      bf[j].q[1] = *(const u32x4*)(p + 16);         // K = 16+hl*8 .. +7
    }
#pragma unroll
    for (int i = 0; i < 4; ++i) {
      Frag16 af;
      const unsigned short* p = &As[(wm * 64 + i * 16 + mn) * 32 + hl * 8];
      af.q[0] = *(const u32x4*)p;
      af.q[1] = *(const u32x4*)(p + 16);
#pragma unroll
      for (int j = 0; j < 2; ++j)
        acc[i][j] = __builtin_amdgcn_wmma_f32_16x16x32_bf16(
            false, af.v, false, bf[j].v, (short)0, acc[i][j], false, false);
    }
  }

  // epilogue: C layout = lanes 0-15 col N=lane, VGPR e -> M=e (lanes>=16: M=e+8)
  const int hl = lane >> 4, nn = lane & 15;
#pragma unroll
  for (int i = 0; i < 4; ++i)
#pragma unroll
    for (int j = 0; j < 2; ++j)
#pragma unroll
      for (int e = 0; e < 8; ++e) {
        int row = bm + wm * 64 + i * 16 + (hl ? e + 8 : e);
        int col = bn + wn * 32 + j * 16 + nn;
        float v = acc[i][j][e];
        if (OUT_F32) {
          ((float*)Cout)[(size_t)row * N + col] = v + (ADD_BIAS ? bias[col] : 0.0f);
        } else {
          ((unsigned short*)Cout)[(size_t)row * N + col] = f2bf(v);
        }
      }
}

// --------------------------------------------------------------- attention --
// One block per (head, window, batch). 256 threads = 8 waves.
// LDS: S[208][208] f32 (aliased by P bf16, row stride 416) | q[208][64] |
//      k[208][64] | vT[64][224]   -> 254976 bytes (dynamic).
__global__ __launch_bounds__(256) void k_win_attn(
    const unsigned short* __restrict__ qkv,   // [B*3600][2304] bf16
    unsigned short* __restrict__ ao) {        // [B*3600][768]  bf16
  const int head = blockIdx.x;                // 0..11
  const int win  = blockIdx.y;                // 0..24
  const int b    = blockIdx.z;                // 0..7
  const int wy = win / 5, wx = win - wy * 5;

  extern __shared__ char smem[];
  float*          Sm = (float*)smem;                               // 208*208 f32
  unsigned short* Pm = (unsigned short*)smem;                      // stride 416
  unsigned short* qs = (unsigned short*)(smem + 173056);           // 208*64
  unsigned short* ks = (unsigned short*)(smem + 173056 + 26624);   // 208*64
  unsigned short* vt = (unsigned short*)(smem + 173056 + 53248);   // 64*224

  const int tid  = threadIdx.x;
  const int lane = tid & 31, wave = tid >> 5;
  const int hl = lane >> 4, mn = lane & 15;

  // ---- gather q,k windows into LDS (zero-fill the H/W padding region) ----
  for (int idx = tid; idx < 2 * 196 * 8; idx += 256) {
    int mat = idx / (196 * 8);
    int rem = idx - mat * 196 * 8;
    int t = rem >> 3, ch = (rem & 7) * 8;
    int ty = t / 14, tx = t - ty * 14;
    int h = wy * 14 + ty, w = wx * 14 + tx;
    u32x4 val = (u32x4){0u, 0u, 0u, 0u};
    if (h < 60 && w < 60) {
      size_t row = (size_t)b * 3600 + h * 60 + w;
      val = *(const u32x4*)(qkv + row * 2304 + mat * 768 + head * 64 + ch);
    }
    unsigned short* dst = mat ? ks : qs;
    *(u32x4*)(dst + t * 64 + ch) = val;
  }
  // ---- gather v transposed: vT[d][t] ----
  for (int idx = tid; idx < 196 * 8; idx += 256) {
    int t = idx >> 3, ch = (idx & 7) * 8;
    int ty = t / 14, tx = t - ty * 14;
    int h = wy * 14 + ty, w = wx * 14 + tx;
    unsigned short tmp[8];
    if (h < 60 && w < 60) {
      size_t row = (size_t)b * 3600 + h * 60 + w;
      u32x4 val = *(const u32x4*)(qkv + row * 2304 + 1536 + head * 64 + ch);
      *(u32x4*)tmp = val;
    } else {
#pragma unroll
      for (int j = 0; j < 8; ++j) tmp[j] = 0;
    }
#pragma unroll
    for (int j = 0; j < 8; ++j) vt[(ch + j) * 224 + t] = tmp[j];
  }
  // ---- zero the Nw->tile padding rows/cols ----
  for (int i = tid; i < 2 * 12 * 64; i += 256) {    // q/k rows 196..207
    int mat = i / 768, r = i - mat * 768;
    (mat ? ks : qs)[196 * 64 + r] = 0;
  }
  for (int i = tid; i < 64 * 28; i += 256) {        // vT cols 196..223
    int d = i / 28, t = 196 + (i - d * 28);
    vt[d * 224 + t] = 0;
  }
  __syncthreads();

  // ---- S = q @ k^T * scale : 13x13 tiles of 16x16, K=64 (2 WMMAs) ----
  for (int tile = wave; tile < 169; tile += 8) {
    int tm = tile / 13, tn = tile - tm * 13;
    v8f c;
#pragma unroll
    for (int e = 0; e < 8; ++e) c[e] = 0.0f;
#pragma unroll
    for (int kc = 0; kc < 64; kc += 32) {
      Frag16 a, bb;
      const unsigned short* pa = &qs[(tm * 16 + mn) * 64 + kc + hl * 8];
      a.q[0] = *(const u32x4*)pa;
      a.q[1] = *(const u32x4*)(pa + 16);
      const unsigned short* pb = &ks[(tn * 16 + mn) * 64 + kc + hl * 8];
      bb.q[0] = *(const u32x4*)pb;
      bb.q[1] = *(const u32x4*)(pb + 16);
      c = __builtin_amdgcn_wmma_f32_16x16x32_bf16(
          false, a.v, false, bb.v, (short)0, c, false, false);
    }
#pragma unroll
    for (int e = 0; e < 8; ++e) {
      int row = tm * 16 + (hl ? e + 8 : e);
      int col = tn * 16 + mn;
      Sm[row * 208 + col] = c[e] * 0.125f;          // scale = hd^-0.5
    }
  }
  __syncthreads();

  // ---- softmax rows; write P bf16 in place over S (stride 416 ushorts).
  //      In-row write offset (2c bytes) always trails read offset (4c). ----
  {
    int r = tid;
    if (r < 196) {
      const float* srow = &Sm[r * 208];
      float m = -1e30f;
      for (int c = 0; c < 196; ++c) m = fmaxf(m, srow[c]);
      float sum = 0.0f;
      for (int c = 0; c < 196; ++c) sum += __expf(srow[c] - m);
      float inv = 1.0f / sum;
      unsigned short* prow = &Pm[r * 416];
      for (int c = 0; c < 224; ++c) {
        float p = (c < 196) ? __expf(srow[c] - m) * inv : 0.0f;
        prow[c] = f2bf(p);
      }
    } else if (r < 208) {
      unsigned short* prow = &Pm[r * 416];
      for (int c = 0; c < 224; ++c) prow[c] = 0;
    }
  }
  __syncthreads();

  // ---- O = P @ v : 13x4 tiles of 16x16, K=224 (7 WMMAs); scatter-store ----
  for (int tile = wave; tile < 52; tile += 8) {
    int tm = tile >> 2, tn = tile & 3;
    v8f c;
#pragma unroll
    for (int e = 0; e < 8; ++e) c[e] = 0.0f;
#pragma unroll
    for (int kc = 0; kc < 224; kc += 32) {
      Frag16 a, bb;
      const unsigned short* pa = &Pm[(tm * 16 + mn) * 416 + kc + hl * 8];
      a.q[0] = *(const u32x4*)pa;
      a.q[1] = *(const u32x4*)(pa + 16);
      const unsigned short* pb = &vt[(tn * 16 + mn) * 224 + kc + hl * 8];
      bb.q[0] = *(const u32x4*)pb;
      bb.q[1] = *(const u32x4*)(pb + 16);
      c = __builtin_amdgcn_wmma_f32_16x16x32_bf16(
          false, a.v, false, bb.v, (short)0, c, false, false);
    }
#pragma unroll
    for (int e = 0; e < 8; ++e) {
      int row = tm * 16 + (hl ? e + 8 : e);
      if (row < 196) {
        int ty = row / 14, tx = row - ty * 14;
        int h = wy * 14 + ty, w = wx * 14 + tx;
        if (h < 60 && w < 60) {                      // crop padding (un-fold)
          size_t orow = (size_t)b * 3600 + h * 60 + w;
          ao[orow * 768 + head * 64 + tn * 16 + mn] = f2bf(c[e]);
        }
      }
    }
  }
}

// ------------------------------------------------------------------ launch --
extern "C" void kernel_launch(void* const* d_in, const int* in_sizes, int n_in,
                              void* d_out, int out_size, void* d_ws, size_t ws_size,
                              hipStream_t stream) {
  const float* x      = (const float*)d_in[0];   // [8,3600,768]
  const float* w_qkv  = (const float*)d_in[1];   // [768,2304]
  const float* w_proj = (const float*)d_in[2];   // [768,768]
  const float* b_proj = (const float*)d_in[3];   // [768]
  // d_in[4]=H, d_in[5]=W are compile-time constants here (60, 60).

  constexpr int M = 8 * 3600;                    // 28800
  constexpr int C = 768;
  constexpr int N1 = 3 * C;                      // 2304

  unsigned short* xb     = (unsigned short*)d_ws;
  unsigned short* wqkvT  = xb     + (size_t)M * C;        //  44.2 MB in
  unsigned short* wprojT = wqkvT  + (size_t)N1 * C;
  unsigned short* qkvb   = wprojT + (size_t)C * C;
  unsigned short* aob    = qkvb   + (size_t)M * N1;       // total ~226 MB

  k_f32_to_bf16<<<2048, 256, 0, stream>>>(x, xb, M * C);
  k_transpose_to_bf16<<<2048, 256, 0, stream>>>(w_qkv, wqkvT, C, N1);
  k_transpose_to_bf16<<<1024, 256, 0, stream>>>(w_proj, wprojT, C, C);

  // qkv = x @ w_qkv  -> bf16
  k_gemm_bf16<false, false><<<dim3(N1 / 128, M / 128), 256, 0, stream>>>(
      xb, wqkvT, nullptr, qkvb, M, N1, C);

  // windowed attention: 12 heads x 25 windows x 8 batches
  k_win_attn<<<dim3(12, 25, 8), 256, 254976, stream>>>(qkvb, aob);

  // out = ao @ w_proj + b_proj -> f32
  k_gemm_bf16<true, true><<<dim3(C / 128, M / 128), 256, 0, stream>>>(
      aob, wprojT, b_proj, (float*)d_out, M, C, C);
}
